// DQGNN_89541478187586
// MI455X (gfx1250) — compile-verified, matrix operand order checked by reading, not run on hardware
//
#include <hip/hip_runtime.h>
#include <hip/hip_bf16.h>

// CDNA5 / gfx1250 implementation.
// Dense matmuls (GraphConv node updates + output MLP) use
// v_wmma_f32_16x16x32_f16 (f16 in, f32 accumulate). Edge aggregation uses
// global_atomic_add_f32 scatter (memory/atomic bound; no matrix op applies).
// Output-MLP weights are staged into LDS once per workgroup via the Tensor
// Data Mover (tensor_load_to_lds, TENSORcnt) when the builtin is available.

typedef __attribute__((ext_vector_type(16))) _Float16 v16h;
typedef __attribute__((ext_vector_type(8)))  float    v8f;

#define BGRAPH   32768
#define NODESPG  54
#define NNODE    (BGRAPH * NODESPG)      // 1,769,472
#define NEDGE    (BGRAPH * 144)          // 4,718,592
#define ROWTILES (BGRAPH / 16)           // 2048

// ---- WMMA fragment K indexing (CDNA5 ISA §7.12.2, 16-bit A 16x32) ----
// VGPR v (v<4): halves (2v + 8*hi, +1); v>=4: (16 + 2(v-4) + 8*hi, +1).
__device__ __forceinline__ int kbase(int v, int hi) {
    return (v < 4) ? (2 * v + 8 * hi) : (16 + 2 * (v - 4) + 8 * hi);
}

// ------------------------------------------------ TDM global->LDS staging --
#if defined(__has_builtin)
#if __has_builtin(__builtin_amdgcn_tensor_load_to_lds)
#define HAVE_TDM 1
#endif
#endif

#ifdef HAVE_TDM
typedef unsigned int u32x4 __attribute__((ext_vector_type(4)));
typedef int          i32x8v __attribute__((ext_vector_type(8)));
typedef int          i32x4v __attribute__((ext_vector_type(4)));

// 1-D contiguous copy: data_size=8B, tile = whole tensor (bytes/8 <= 65535).
// D# layout per CDNA5 ISA.
// This toolchain's builtin takes 6 args (clang-23 / therock form):
//   (u32x4 g0, i32x8 g1, i32x4 g2, i32x4 g3, i32x8 extra, i32 cpol).
__device__ __forceinline__ void tdm_copy_to_lds(const void* gsrc, void* lds,
                                                unsigned bytes) {
    unsigned nunits = bytes >> 3;                       // 8-byte units
    unsigned long long ga = (unsigned long long)gsrc;
    unsigned la = (unsigned)(unsigned long long)lds;    // LDS byte offset
    u32x4 g0;
    g0[0] = 1u;                                   // count=1, user descriptor
    g0[1] = la;                                   // lds_addr
    g0[2] = (unsigned)ga;                         // global_addr[31:0]
    g0[3] = (unsigned)(ga >> 32) | (2u << 30);    // global_addr[56:32] | type=2
    i32x8v g1;
    g1[0] = (int)(3u << 16);                      // wg_mask=0, data_size=3 (8B)
    g1[1] = (int)((nunits & 0xFFFFu) << 16);      // tensor_dim0[15:0] @ b63:48
    g1[2] = (int)((nunits >> 16) | (1u << 16));   // tensor_dim0[31:16] | td1=1
    g1[3] = (int)(nunits << 16);                  // tile_dim0 @ b127:112
    g1[4] = 0;                                    // tile_dim1=0, tile_dim2=0
    g1[5] = (int)nunits;                          // tensor_dim0_stride[31:0]
    g1[6] = 0;
    g1[7] = 0;
    i32x4v z4 = {0, 0, 0, 0};
    i32x8v z8 = {0, 0, 0, 0, 0, 0, 0, 0};
    __builtin_amdgcn_tensor_load_to_lds(g0, g1, z4, z4, z8, 0);
}
#endif

// ---------------------------------------------------------------- zero ----
__global__ void zero_f4(float4* p, size_t n4) {
    size_t i = blockIdx.x * (size_t)blockDim.x + threadIdx.x;
    size_t stride = (size_t)gridDim.x * blockDim.x;
    for (; i < n4; i += stride) p[i] = make_float4(0.f, 0.f, 0.f, 0.f);
}

// --------------------------------------------------- edge scatter (atomic) -
// conv1: 4 threads/edge, each handles 4 of 16 channels (float4 gather).
__global__ void scatter1(const float* __restrict__ x,
                         const long long* __restrict__ ei,
                         const float* __restrict__ ew,
                         float* __restrict__ aggr) {
    size_t t = blockIdx.x * (size_t)blockDim.x + threadIdx.x;
    size_t e = t >> 2;
    int p = (int)(t & 3);
    if (e >= (size_t)NEDGE) return;
    size_t src = (size_t)ei[e];
    size_t dst = (size_t)ei[(size_t)NEDGE + e];
    float w = ew[e];
    float4 v = *reinterpret_cast<const float4*>(x + src * 16 + p * 4);
    float* a = aggr + dst * 16 + p * 4;
    atomicAdd(a + 0, v.x * w);
    atomicAdd(a + 1, v.y * w);
    atomicAdd(a + 2, v.z * w);
    atomicAdd(a + 3, v.w * w);
}

// conv2: 8 threads/edge over 32 channels.
__global__ void scatter2(const float* __restrict__ h1,
                         const long long* __restrict__ ei,
                         const float* __restrict__ ew,
                         float* __restrict__ aggr) {
    size_t t = blockIdx.x * (size_t)blockDim.x + threadIdx.x;
    size_t e = t >> 3;
    int p = (int)(t & 7);
    if (e >= (size_t)NEDGE) return;
    size_t src = (size_t)ei[e];
    size_t dst = (size_t)ei[(size_t)NEDGE + e];
    float w = ew[e];
    float4 v = *reinterpret_cast<const float4*>(h1 + src * 32 + p * 4);
    float* a = aggr + dst * 32 + p * 4;
    atomicAdd(a + 0, v.x * w);
    atomicAdd(a + 1, v.y * w);
    atomicAdd(a + 2, v.z * w);
    atomicAdd(a + 3, v.w * w);
}

// --------------------------------------------------------- weight packing --
// Pack row-major f32 W[Kreal x Nreal] into per-(ktile,ntile) B fragments:
// lane l holds column n=l&15, K halves per kbase(); one v16h (32B) per lane.
__global__ void pack_B(const float* __restrict__ W, v16h* __restrict__ Bp,
                       int Kreal, int Nreal, int NT, int numTiles) {
    int t = blockIdx.x * blockDim.x + threadIdx.x;
    if (t >= numTiles * 32) return;
    int lane = t & 31, tile = t >> 5;
    int kt = tile / NT, nt = tile % NT;
    int n = nt * 16 + (lane & 15), hi = lane >> 4;
    v16h f;
#pragma unroll
    for (int v = 0; v < 8; v++) {
        int k = kt * 32 + kbase(v, hi);
        float a0 = (k     < Kreal && n < Nreal) ? W[(size_t)k       * Nreal + n] : 0.f;
        float a1 = (k + 1 < Kreal && n < Nreal) ? W[(size_t)(k + 1) * Nreal + n] : 0.f;
        f[2 * v]     = (_Float16)a0;
        f[2 * v + 1] = (_Float16)a1;
    }
    Bp[t] = f;
}

// Stacked B = [Wa; Wb], each Ka x Nreal (GraphConv: [W_rel; W_root]).
__global__ void pack_Bstack(const float* __restrict__ Wa,
                            const float* __restrict__ Wb,
                            v16h* __restrict__ Bp,
                            int Ka, int Nreal, int NT, int numTiles) {
    int t = blockIdx.x * blockDim.x + threadIdx.x;
    if (t >= numTiles * 32) return;
    int lane = t & 31, tile = t >> 5;
    int kt = tile / NT, nt = tile % NT;
    int n = nt * 16 + (lane & 15), hi = lane >> 4;
    v16h f;
#pragma unroll
    for (int v = 0; v < 8; v++) {
        int k = kt * 32 + kbase(v, hi);
        float a0 = 0.f, a1 = 0.f;
        if (n < Nreal) {
            int k0 = k, k1 = k + 1;
            a0 = (k0 < Ka) ? Wa[(size_t)k0 * Nreal + n]
                           : ((k0 < 2 * Ka) ? Wb[(size_t)(k0 - Ka) * Nreal + n] : 0.f);
            a1 = (k1 < Ka) ? Wa[(size_t)k1 * Nreal + n]
                           : ((k1 < 2 * Ka) ? Wb[(size_t)(k1 - Ka) * Nreal + n] : 0.f);
        }
        f[2 * v]     = (_Float16)a0;
        f[2 * v + 1] = (_Float16)a1;
    }
    Bp[t] = f;
}

// ----------------------------------------------- GraphConv1 node update ----
// h1 = relu([aggr1 | x] @ [W1_rel; W1_root] + b1)   K=32 exact, Nout=32.
// One wave per 16 nodes: 1 A fragment, 2 WMMAs.
__global__ __launch_bounds__(256) void conv_node1(
        const float* __restrict__ aggr, const float* __restrict__ x,
        const v16h* __restrict__ Bp, const float* __restrict__ bias,
        float* __restrict__ h1) {
    int gid = blockIdx.x * blockDim.x + threadIdx.x;
    int wave = gid >> 5, lane = gid & 31;
    int n = lane & 15, hi = lane >> 4;
    size_t node = (size_t)wave * 16 + n;   // A-row for this lane
    v16h a;
#pragma unroll
    for (int v = 0; v < 8; v++) {
        int k = kbase(v, hi);
        const float* sp = (k < 16) ? (aggr + node * 16 + k)
                                   : (x    + node * 16 + (k - 16));
        a[2 * v]     = (_Float16)sp[0];
        a[2 * v + 1] = (_Float16)sp[1];
    }
    v8f acc0 = {}, acc1 = {};
    v16h b0 = Bp[lane], b1v = Bp[32 + lane];
    acc0 = __builtin_amdgcn_wmma_f32_16x16x32_f16(false, a, false, b0,  (short)0, acc0, false, false);
    acc1 = __builtin_amdgcn_wmma_f32_16x16x32_f16(false, a, false, b1v, (short)0, acc1, false, false);
    float bv0 = bias[n], bv1 = bias[16 + n];
#pragma unroll
    for (int r = 0; r < 8; r++) {
        int m = r + 8 * hi;
        size_t nd = (size_t)wave * 16 + m;
        float v0 = acc0[r] + bv0; v0 = v0 > 0.f ? v0 : 0.f;
        float v1 = acc1[r] + bv1; v1 = v1 > 0.f ? v1 : 0.f;
        h1[nd * 32 + n]      = v0;
        h1[nd * 32 + 16 + n] = v1;
    }
}

// ----------------------------------------------- GraphConv2 node update ----
// h2 = relu([aggr2 | h1] @ [W2_rel; W2_root] + b2)  K=64 (2 ktiles), Nout=4.
__global__ __launch_bounds__(256) void conv_node2(
        const float* __restrict__ aggr, const float* __restrict__ h1,
        const v16h* __restrict__ Bp, const float* __restrict__ bias,
        float* __restrict__ h2) {
    int gid = blockIdx.x * blockDim.x + threadIdx.x;
    int wave = gid >> 5, lane = gid & 31;
    int n = lane & 15, hi = lane >> 4;
    size_t node = (size_t)wave * 16 + n;
    v8f acc = {};
#pragma unroll
    for (int kt = 0; kt < 2; kt++) {
        const float* src = (kt == 0) ? aggr : h1;
        v16h a;
#pragma unroll
        for (int v = 0; v < 8; v++) {
            int k = kbase(v, hi);
            const float* sp = src + node * 32 + k;
            a[2 * v]     = (_Float16)sp[0];
            a[2 * v + 1] = (_Float16)sp[1];
        }
        v16h b = Bp[kt * 32 + lane];
        acc = __builtin_amdgcn_wmma_f32_16x16x32_f16(false, a, false, b, (short)0, acc, false, false);
    }
    if (n < 4) {
        float bv = bias[n];
#pragma unroll
        for (int r = 0; r < 8; r++) {
            int m = r + 8 * hi;
            size_t nd = (size_t)wave * 16 + m;
            float v = acc[r] + bv;
            h2[nd * 4 + n] = v > 0.f ? v : 0.f;
        }
    }
}

// ----------------------------------------------------------- global MLP ----
// 16 -> 8 -> 8 -> 16, relu each. One thread per graph; weights in LDS.
__global__ __launch_bounds__(256) void gmlp(
        const float* __restrict__ gin,
        const float* __restrict__ Wg1, const float* __restrict__ bg1,
        const float* __restrict__ Wg2, const float* __restrict__ bg2,
        const float* __restrict__ Wg3, const float* __restrict__ bg3,
        float* __restrict__ gout) {
    __shared__ float s[352];  // w1[128] w2[64]@128 w3[128]@192 b1@320 b2@328 b3@336
    int tid = threadIdx.x;
    if (tid < 128) s[tid] = Wg1[tid];
    if (tid < 64)  s[128 + tid] = Wg2[tid];
    if (tid < 128) s[192 + tid] = Wg3[tid];
    if (tid < 8)   s[320 + tid] = bg1[tid];
    if (tid < 8)   s[328 + tid] = bg2[tid];
    if (tid < 16)  s[336 + tid] = bg3[tid];
    __syncthreads();
    size_t b = blockIdx.x * 256 + tid;
    float in[16];
#pragma unroll
    for (int i = 0; i < 16; i++) in[i] = gin[b * 16 + i];
    float ha[8];
#pragma unroll
    for (int j = 0; j < 8; j++) {
        float acc = s[320 + j];
        for (int i = 0; i < 16; i++) acc += in[i] * s[i * 8 + j];
        ha[j] = acc > 0.f ? acc : 0.f;
    }
    float hb[8];
#pragma unroll
    for (int j = 0; j < 8; j++) {
        float acc = s[328 + j];
        for (int i = 0; i < 8; i++) acc += ha[i] * s[128 + i * 8 + j];
        hb[j] = acc > 0.f ? acc : 0.f;
    }
#pragma unroll
    for (int j = 0; j < 16; j++) {
        float acc = s[336 + j];
        for (int i = 0; i < 8; i++) acc += hb[i] * s[192 + i * 16 + j];
        gout[b * 16 + j] = acc > 0.f ? acc : 0.f;
    }
}

// ------------------------------------------- MLP input assembly (packed) ---
// in-row b = [embeds (h2 flat, 216) | g (16) | 0-pad to 256], written
// directly in A-fragment-packed f16 layout: 8 ktiles x 32 lanes x 32B.
__device__ __forceinline__ float fetch_in(const float* h2, const float* g,
                                          size_t row, int k) {
    if (k < 216) return h2[row * 216 + k];
    if (k < 232) return g[row * 16 + (k - 216)];
    return 0.f;
}

__global__ __launch_bounds__(256) void assemble_in(
        const float* __restrict__ h2, const float* __restrict__ g,
        v16h* __restrict__ Ap) {
    int t = blockIdx.x * blockDim.x + threadIdx.x;  // ROWTILES*8*32 threads
    int lane = t & 31;
    int kt = (t >> 5) & 7;
    int rt = t >> 8;
    int m = lane & 15, hi = lane >> 4;
    size_t row = (size_t)rt * 16 + m;
    v16h f;
#pragma unroll
    for (int v = 0; v < 8; v++) {
        int k = kt * 32 + kbase(v, hi);
        f[2 * v]     = (_Float16)fetch_in(h2, g, row, k);
        f[2 * v + 1] = (_Float16)fetch_in(h2, g, row, k + 1);
    }
    Ap[((size_t)rt * 8 + kt) * 32 + lane] = f;
}

// ------------------------------------------------------ WMMA MLP layer -----
// One wave per 16-row tile, 8 waves/block. The layer's full packed-B block
// is staged into LDS once per workgroup (TDM DMA when available), then each
// wave reads B fragments with ds_read_b128. Non-final layers relu + convert
// to f16 via a per-wave LDS bounce and emit A-fragment-packed activations.
template <int KT, int NT, bool FINAL>
__global__ __launch_bounds__(256) void mlp_layer(
        const v16h* __restrict__ Ap, const v16h* __restrict__ Bp,
        const float* __restrict__ bias,
        v16h* __restrict__ outPacked, float* __restrict__ outFinal) {
    constexpr int NCOLS = NT * 16;
    constexpr int BTILES = KT * NT;
    constexpr int TILE_ELEMS = FINAL ? 1 : (8 * 16 * NCOLS);
    __shared__ _Float16 tile[TILE_ELEMS];
    __shared__ __attribute__((aligned(16))) _Float16 sB[BTILES * 512]; // BTILES KB

    // --- stage packed B into LDS, shared by all 8 waves ---
#ifdef HAVE_TDM
    if (threadIdx.x < 32) {                 // wave 0 issues the DMA
        tdm_copy_to_lds(Bp, sB, BTILES * 1024);
        __builtin_amdgcn_s_wait_tensorcnt(0);
    }
#else
    for (int i = threadIdx.x; i < BTILES * 64; i += 256)
        ((float4*)sB)[i] = ((const float4*)Bp)[i];
#endif
    __syncthreads();

    int lane = threadIdx.x & 31;
    int w = threadIdx.x >> 5;
    int rt = blockIdx.x * 8 + w;
    int n = lane & 15, hi = lane >> 4;
    const v16h* sBf = (const v16h*)sB;

    v8f acc[NT] = {};
#pragma unroll
    for (int kt = 0; kt < KT; kt++) {
        v16h a = Ap[((size_t)rt * KT + kt) * 32 + lane];
#pragma unroll
        for (int nt = 0; nt < NT; nt++) {
            v16h b = sBf[(kt * NT + nt) * 32 + lane];
            acc[nt] = __builtin_amdgcn_wmma_f32_16x16x32_f16(
                false, a, false, b, (short)0, acc[nt], false, false);
        }
    }

    if constexpr (FINAL) {
#pragma unroll
        for (int nt = 0; nt < NT; nt++) {
            float bv = bias[nt * 16 + n];
#pragma unroll
            for (int r = 0; r < 8; r++) {
                int m = r + 8 * hi;
                outFinal[((size_t)rt * 16 + m) * NCOLS + nt * 16 + n] = acc[nt][r] + bv;
            }
        }
    } else {
        _Float16* tp = &tile[w * 16 * NCOLS];
#pragma unroll
        for (int nt = 0; nt < NT; nt++) {
            float bv = bias[nt * 16 + n];
#pragma unroll
            for (int r = 0; r < 8; r++) {
                int m = r + 8 * hi;
                float v = acc[nt][r] + bv;
                v = v > 0.f ? v : 0.f;
                tp[m * NCOLS + nt * 16 + n] = (_Float16)v;
            }
        }
        __syncthreads();
        constexpr int KTO = NCOLS / 32;
        const _Float16* rowp = &tile[w * 16 * NCOLS + n * NCOLS];  // row m = n
#pragma unroll
        for (int kt2 = 0; kt2 < KTO; kt2++) {
            v16h f;
#pragma unroll
            for (int v = 0; v < 8; v++) {
                int k = kt2 * 32 + kbase(v, hi);
                f[2 * v]     = rowp[k];
                f[2 * v + 1] = rowp[k + 1];
            }
            outPacked[((size_t)rt * KTO + kt2) * 32 + lane] = f;
        }
    }
}

// ============================================================ launcher =====
extern "C" void kernel_launch(void* const* d_in, const int* in_sizes, int n_in,
                              void* d_out, int out_size, void* d_ws, size_t ws_size,
                              hipStream_t stream) {
    const float*     x  = (const float*)d_in[0];
    const long long* ei = (const long long*)d_in[1];
    const float*     ew = (const float*)d_in[2];
    const float*     gf = (const float*)d_in[3];
    // d_in[4] = isTrain (0 at inference; dropout is identity)
    const float* W1_rel = (const float*)d_in[5];
    const float* b1     = (const float*)d_in[6];
    const float* W1_root= (const float*)d_in[7];
    const float* W2_rel = (const float*)d_in[8];
    const float* b2     = (const float*)d_in[9];
    const float* W2_root= (const float*)d_in[10];
    const float* Wg1 = (const float*)d_in[11]; const float* bg1 = (const float*)d_in[12];
    const float* Wg2 = (const float*)d_in[13]; const float* bg2 = (const float*)d_in[14];
    const float* Wg3 = (const float*)d_in[15]; const float* bg3 = (const float*)d_in[16];
    const float* Wo1 = (const float*)d_in[17]; const float* bo1 = (const float*)d_in[18];
    const float* Wo2 = (const float*)d_in[19]; const float* bo2 = (const float*)d_in[20];
    const float* Wo3 = (const float*)d_in[21]; const float* bo3 = (const float*)d_in[22];

    char* ws = (char*)d_ws;
    size_t off = 0;
    auto alloc = [&](size_t bytes) -> void* {
        void* p = ws + off;
        off += (bytes + 255) & ~(size_t)255;
        return p;
    };
    // aggr1 (N x 16) aliases the front of aggr2 (N x 32); re-zeroed between.
    float* aggr2 = (float*)alloc((size_t)NNODE * 32 * 4);
    float* aggr1 = aggr2;
    float* h1    = (float*)alloc((size_t)NNODE * 32 * 4);
    float* h2    = (float*)alloc((size_t)NNODE * 4 * 4);
    float* gbuf  = (float*)alloc((size_t)BGRAPH * 16 * 4);
    v16h* inP  = (v16h*)alloc((size_t)ROWTILES * 8 * 1024);
    v16h* act1 = (v16h*)alloc((size_t)ROWTILES * 4 * 1024);
    v16h* act2 = (v16h*)alloc((size_t)ROWTILES * 4 * 1024);
    v16h* B1p  = (v16h*)alloc(64 * 1024);
    v16h* B2p  = (v16h*)alloc(32 * 1024);
    v16h* B3p  = (v16h*)alloc(16 * 1024);
    v16h* C1p  = (v16h*)alloc(2 * 1024);
    v16h* C2p  = (v16h*)alloc(2 * 1024);

    // --- weight packing (recomputed every call; deterministic) ---
    pack_Bstack<<<1, 64, 0, stream>>>(W1_rel, W1_root, C1p, 16, 32, 2, 2);
    pack_Bstack<<<1, 64, 0, stream>>>(W2_rel, W2_root, C2p, 32, 4, 1, 2);
    pack_B<<<8, 256, 0, stream>>>(Wo1, B1p, 232, 128, 8, 64);
    pack_B<<<4, 256, 0, stream>>>(Wo2, B2p, 128, 128, 8, 32);
    pack_B<<<2, 256, 0, stream>>>(Wo3, B3p, 128, 64, 4, 16);

    // --- GraphConv 1 ---
    zero_f4<<<4096, 256, 0, stream>>>((float4*)aggr1, (size_t)NNODE * 16 / 4);
    scatter1<<<(int)(((size_t)NEDGE * 4) / 256), 256, 0, stream>>>(x, ei, ew, aggr1);
    conv_node1<<<(NNODE / 16) * 32 / 256, 256, 0, stream>>>(aggr1, x, C1p, b1, h1);

    // --- GraphConv 2 (re-zero the aliased aggregation buffer first) ---
    zero_f4<<<4096, 256, 0, stream>>>((float4*)aggr2, (size_t)NNODE * 32 / 4);
    scatter2<<<(int)(((size_t)NEDGE * 8) / 256), 256, 0, stream>>>(h1, ei, ew, aggr2);
    conv_node2<<<(NNODE / 16) * 32 / 256, 256, 0, stream>>>(aggr2, h1, C2p, b2, h2);

    // --- global-feature MLP ---
    gmlp<<<BGRAPH / 256, 256, 0, stream>>>(gf, Wg1, bg1, Wg2, bg2, Wg3, bg3, gbuf);

    // --- output MLP (WMMA, f16 activations / f32 accumulate) ---
    assemble_in<<<(ROWTILES * 8 * 32) / 256, 256, 0, stream>>>(h2, gbuf, inP);
    mlp_layer<8, 8, false><<<ROWTILES / 8, 256, 0, stream>>>(inP,  B1p, bo1, act1, nullptr);
    mlp_layer<4, 8, false><<<ROWTILES / 8, 256, 0, stream>>>(act1, B2p, bo2, act2, nullptr);
    mlp_layer<4, 4, true ><<<ROWTILES / 8, 256, 0, stream>>>(act2, B3p, bo3, nullptr, (float*)d_out);

    (void)in_sizes; (void)n_in; (void)out_size; (void)ws_size;
}
